// TextLSTM_66563403153582
// MI455X (gfx1250) — matrix operand order repaired and checked by reading, not compiled, not run on hardware
//
#include <hip/hip_runtime.h>
#include <hip/hip_bf16.h>
#include <stdint.h>

// ---------------- problem dims ----------------
constexpr int Bb = 64, Tt = 512, Ee = 256, Hh = 512, Vv = 32000;
constexpr int NBLK = 32, TPB = 256;          // persistent scan kernel: 32 WGs x 8 wave32
constexpr int WAVES = NBLK * (TPB / 32);     // 256 waves
constexpr int BH = Bb * Hh;                  // 32768

typedef __bf16 bf16_t;
typedef bf16_t v16bf __attribute__((ext_vector_type(16)));
typedef float  v8f   __attribute__((ext_vector_type(8)));

union Frag { v16bf v; uint4 q[2]; };

__device__ __forceinline__ v8f wmma_bf16(v16bf a, v16bf b, v8f c) {
  // (neg_a, A, neg_b, B, c_mod, C, reuse_a, reuse_b)
  return __builtin_amdgcn_wmma_f32_16x16x32_bf16(false, a, false, b, (short)0, c, false, false);
}

__device__ __forceinline__ unsigned short f2bf(float x) {  // round-to-nearest-even
  union { float f; unsigned u; } v; v.f = x;
  unsigned r = v.u + 0x7fffu + ((v.u >> 16) & 1u);
  return (unsigned short)(r >> 16);
}

__device__ __forceinline__ float sigm(float x) { return 1.0f / (1.0f + __expf(-x)); }

// ---- WMMA fragment loaders ----
// A-fragment (ISA 7.12.2, 16-bit A 16x32): lanes 0-15 take K 0-7 / 16-23,
// lanes 16-31 take K 8-15 / 24-31 -> two b128 loads from a row-major row.
__device__ __forceinline__ void load_a(Frag& a, const unsigned short* Arow, int kt, int kh) {
  const uint4* ap = (const uint4*)(Arow + kt * 32 + 8 * kh);
  a.q[0] = ap[0];           // K base+0..7   (pairs in dwords)
  a.q[1] = ap[2];           // K base+16..23
}
// B-fragment from packed weights: per tile, lane l holds 16 contiguous bf16.
__device__ __forceinline__ void load_b(Frag& b, const unsigned short* Bp, int kt, int lane) {
  const uint4* bp = (const uint4*)(Bp + (size_t)kt * 512 + lane * 16);
  b.q[0] = bp[0];
  b.q[1] = bp[1];
}

// TWO M-tiles sharing one B fragment (B from L2 loaded once; A from LDS twice).
// #pragma unroll 2 keeps the live fragment set small -> no scratch spills.
template<int KT>
__device__ __forceinline__ void mm_2m(const unsigned short* __restrict__ A0,
                                      const unsigned short* __restrict__ A1,
                                      const unsigned short* __restrict__ Bp,
                                      int lane, v8f& acc0, v8f& acc1) {
  const int kh = lane >> 4;
  #pragma unroll 2
  for (int kt = 0; kt < KT; ++kt) {
    Frag b, a0, a1;
    load_b(b, Bp, kt, lane);
    load_a(a0, A0, kt, kh);
    load_a(a1, A1, kt, kh);
    acc0 = wmma_bf16(a0.v, b.v, acc0);   // independent D chains -> XDL pipelining
    acc1 = wmma_bf16(a1.v, b.v, acc1);
  }
}

// FOUR M-tiles sharing one B fragment (head: stream W_out exactly once).
template<int KT>
__device__ __forceinline__ void mm_4m(const unsigned short* __restrict__ Abase, int lda,
                                      const unsigned short* __restrict__ Bp,
                                      int lane, v8f& acc0, v8f& acc1, v8f& acc2, v8f& acc3) {
  const int kh = lane >> 4;
  const int m_row = lane & 15;
  const unsigned short* A0 = Abase + (size_t)m_row * lda;
  #pragma unroll 2
  for (int kt = 0; kt < KT; ++kt) {
    Frag b, a0, a1, a2, a3;
    load_b(b, Bp, kt, lane);
    load_a(a0, A0, kt, kh);
    load_a(a1, A0 + 16 * lda, kt, kh);
    load_a(a2, A0 + 32 * lda, kt, kh);
    load_a(a3, A0 + 48 * lda, kt, kh);
    acc0 = wmma_bf16(a0.v, b.v, acc0);
    acc1 = wmma_bf16(a1.v, b.v, acc1);
    acc2 = wmma_bf16(a2.v, b.v, acc2);
    acc3 = wmma_bf16(a3.v, b.v, acc3);
  }
}

// D layout (ISA 7.12.2): VGPR i -> M = i + 8*(lane/16), N = lane%16
__device__ __forceinline__ void store_tile(float* base /*+m0*ldc+n0*/, int ldc, int lane, v8f acc) {
  const int n = lane & 15, kh = lane >> 4;
  #pragma unroll
  for (int i = 0; i < 8; ++i) base[(size_t)(kh * 8 + i) * ldc + n] = acc[i];
}

__device__ __forceinline__ void grid_barrier(unsigned* bar) {
  __syncthreads();
  __threadfence();
  if (threadIdx.x == 0) {
    unsigned old = atomicAdd(bar, 1u);
    unsigned target = (old / NBLK + 1u) * NBLK;
    while (__hip_atomic_load(bar, __ATOMIC_ACQUIRE, __HIP_MEMORY_SCOPE_AGENT) < target)
      __builtin_amdgcn_s_sleep(1);
  }
  __syncthreads();
  __threadfence();
}

// Stage a [64,512] bf16 matrix (64 KB) from global into LDS, whole block.
__device__ __forceinline__ void stage_bh(unsigned short* dst_sh, const unsigned short* src, int tid) {
  const uint4* s = (const uint4*)src;
  uint4* d = (uint4*)dst_sh;
  #pragma unroll 4
  for (int i = tid; i < BH / 8; i += TPB) d[i] = s[i];
}

// Stage x_t [64,256] bf16 (32 KB) rows (stride T*E in global) into LDS.
__device__ __forceinline__ void stage_x(unsigned short* dst_sh, const unsigned short* Xe,
                                        int t, int tid) {
  for (int i = tid; i < Bb * Ee / 8; i += TPB) {   // 2048 uint4
    int b = i >> 5;                                 // 32 uint4 per row
    int q = i & 31;
    ((uint4*)dst_sh)[i] = ((const uint4*)(Xe + ((size_t)b * Tt + t) * Ee))[q];
  }
}

// ---------------- setup kernels ----------------
__global__ void k_init(unsigned short* hbf, unsigned short* h1bf, float* c, float* c1,
                       unsigned* bar) {
  int i = blockIdx.x * blockDim.x + threadIdx.x;
  if (i == 0) *bar = 0u;
  if (i < BH) { hbf[i] = 0x3f80; h1bf[i] = 0x3f80; c[i] = 1.0f; c1[i] = 1.0f; } // states init to ONES
}

__global__ void k_embed(const int* __restrict__ X, const float* __restrict__ emb,
                        unsigned short* __restrict__ Xe) {
  const size_t total = (size_t)Bb * Tt * Ee;
  for (size_t i = (size_t)blockIdx.x * blockDim.x + threadIdx.x; i < total;
       i += (size_t)gridDim.x * blockDim.x) {
    size_t bt = i >> 8;                   // / Ee
    int e = (int)(i & (Ee - 1));
    int tok = X[bt];
    Xe[i] = f2bf(emb[(size_t)tok * Ee + e]);
  }
}

// Pack f32 [K,N] row-major weight into WMMA-B bf16 tiles:
// tile = nt*(K/32)+kt ; per tile: lane l holds N=l&15, K = kt*32 + (l>>4)*8 + {0..7, 16..23},
// contiguous per lane -> 2x b128 loads in the GEMM.
__global__ void k_pack_b(const float* __restrict__ src, unsigned short* __restrict__ dst,
                         int K, int N) {
  const int Kt = K >> 5;
  const size_t total = (size_t)K * N;
  for (size_t i = (size_t)blockIdx.x * blockDim.x + threadIdx.x; i < total;
       i += (size_t)gridDim.x * blockDim.x) {
    size_t tile = i >> 9;                 // / 512
    int within = (int)(i & 511);
    int lane = within >> 4;
    int e = within & 15;
    int kt = (int)(tile % Kt);
    size_t nt = tile / Kt;
    int n = (int)(nt * 16 + (lane & 15));
    int k = kt * 32 + (lane >> 4) * 8 + (e & 7) + (e >> 3) * 16;
    dst[i] = f2bf(src[(size_t)k * N + n]);
  }
}

// ---------------- persistent LSTM scan ----------------
// Dynamic LDS: [0,64K) = shH (h), [64K,128K) = shH1 (h1) aliased with shX (x_t).
__global__ __launch_bounds__(TPB) void k_scan(
    const unsigned short* __restrict__ Xe,
    unsigned short* __restrict__ hbf, unsigned short* __restrict__ h1bf,
    float* __restrict__ c, float* __restrict__ c1,
    float* __restrict__ gA, float* __restrict__ gB,
    const unsigned short* __restrict__ pU0, const unsigned short* __restrict__ pV0,
    const unsigned short* __restrict__ pU1, const unsigned short* __restrict__ pV1,
    const float* __restrict__ b_f, const float* __restrict__ b_i, const float* __restrict__ b_g,
    const float* __restrict__ b_f1, const float* __restrict__ b_i1, const float* __restrict__ b_g1,
    const float* __restrict__ b_o1,
    unsigned* bar) {
  extern __shared__ unsigned short smem[];
  unsigned short* shH  = smem;            // 64 KB
  unsigned short* shH1 = smem + BH;       // 64 KB (phase B) / x_t (phase A)

  const int tid = threadIdx.x;
  const int lane = tid & 31;
  const int gw = blockIdx.x * (TPB / 32) + (tid >> 5);  // global wave id, 0..255
  const int m_row = lane & 15;
  // one job per wave: gate(4) x nt(32) x mt-pair(2) == 256 waves.
  // Two M-tiles share each B (weight) fragment -> L2 weight traffic halved.
  const int gate = gw >> 6, rem = gw & 63, nt = rem >> 1, mt0 = (rem & 1) * 2;
  const size_t rowA = (size_t)(mt0 * 16 + m_row);

  for (int t = 0; t < Tt; ++t) {
    // ---- stage h and x_t into LDS, then layer-0 gates: gA = x_t@U + h@V ----
    stage_bh(shH, hbf, tid);
    stage_x(shH1, Xe, t, tid);
    __syncthreads();
    {
      v8f acc0 = {}, acc1 = {};
      const unsigned short* ax = shH1 + rowA * Ee;
      mm_2m<8>(ax, ax + 16 * Ee, pU0 + (size_t)gate * 131072 + (size_t)nt * 4096,
               lane, acc0, acc1);
      const unsigned short* ah = shH + rowA * Hh;
      mm_2m<16>(ah, ah + 16 * Hh, pV0 + (size_t)gate * 262144 + (size_t)nt * 8192,
                lane, acc0, acc1);
      float* dst = gA + (size_t)gate * BH + (size_t)nt * 16;
      store_tile(dst + (size_t)(mt0 * 16) * Hh, Hh, lane, acc0);
      store_tile(dst + (size_t)((mt0 + 1) * 16) * Hh, Hh, lane, acc1);
    }
    grid_barrier(bar);
    // ---- elementwise layer 0 (note: reference's o-gate uses V_o1/b_o1) ----
    for (int e = blockIdx.x * TPB + tid; e < BH; e += NBLK * TPB) {
      const int hid = e & (Hh - 1);
      float f = sigm(gA[e] + b_f[hid]);
      float i = sigm(gA[BH + e] + b_i[hid]);
      float g = tanhf(gA[2 * BH + e] + b_g[hid]);
      float o = sigm(gA[3 * BH + e] + b_o1[hid]);
      float cv = f * c[e] + i * g;
      c[e] = cv;
      hbf[e] = f2bf(o * tanhf(cv));
    }
    grid_barrier(bar);
    // ---- stage new h and h1, then layer-1 gates: gB = h@U1 + h1@V1 ----
    stage_bh(shH, hbf, tid);
    stage_bh(shH1, h1bf, tid);
    __syncthreads();
    {
      v8f acc0 = {}, acc1 = {};
      const unsigned short* ah = shH + rowA * Hh;
      mm_2m<16>(ah, ah + 16 * Hh, pU1 + (size_t)gate * 262144 + (size_t)nt * 8192,
                lane, acc0, acc1);
      const unsigned short* a1 = shH1 + rowA * Hh;
      mm_2m<16>(a1, a1 + 16 * Hh, pV1 + (size_t)gate * 262144 + (size_t)nt * 8192,
                lane, acc0, acc1);
      float* dst = gB + (size_t)gate * BH + (size_t)nt * 16;
      store_tile(dst + (size_t)(mt0 * 16) * Hh, Hh, lane, acc0);
      store_tile(dst + (size_t)((mt0 + 1) * 16) * Hh, Hh, lane, acc1);
    }
    grid_barrier(bar);
    // ---- elementwise layer 1 ----
    for (int e = blockIdx.x * TPB + tid; e < BH; e += NBLK * TPB) {
      const int hid = e & (Hh - 1);
      float f = sigm(gB[e] + b_f1[hid]);
      float i = sigm(gB[BH + e] + b_i1[hid]);
      float g = tanhf(gB[2 * BH + e] + b_g1[hid]);
      float o = sigm(gB[3 * BH + e] + b_o1[hid]);
      float cv = f * c1[e] + i * g;
      c1[e] = cv;
      h1bf[e] = f2bf(o * tanhf(cv));
    }
    grid_barrier(bar);
  }
}

// ---------------- output head: out = 2*(h1 @ W_out) + b_out ----------------
// One wave per N-tile computing all 4 M-tiles: packed W_out streamed exactly once.
__global__ __launch_bounds__(256) void k_head(const unsigned short* __restrict__ h1bf,
                                              const unsigned short* __restrict__ pW,
                                              const float* __restrict__ b_out,
                                              float* __restrict__ out) {
  extern __shared__ unsigned short smem[];
  const int tid = threadIdx.x;
  stage_bh(smem, h1bf, tid);
  __syncthreads();

  const int lane = tid & 31;
  const int nt = blockIdx.x * (256 / 32) + (tid >> 5);   // 2000 jobs
  if (nt >= Vv / 16) return;
  v8f acc0 = {}, acc1 = {}, acc2 = {}, acc3 = {};
  mm_4m<16>(smem, Hh, pW + (size_t)nt * 8192, lane, acc0, acc1, acc2, acc3);
  const int kh = lane >> 4;
  const int n = nt * 16 + (lane & 15);
  const float bo = b_out[n];
  #pragma unroll
  for (int i = 0; i < 8; ++i) {
    const int mr = kh * 8 + i;
    out[(size_t)(mr) * Vv + n]      = 2.0f * acc0[i] + bo;
    out[(size_t)(mr + 16) * Vv + n] = 2.0f * acc1[i] + bo;
    out[(size_t)(mr + 32) * Vv + n] = 2.0f * acc2[i] + bo;
    out[(size_t)(mr + 48) * Vv + n] = 2.0f * acc3[i] + bo;
  }
}

// ---------------- host orchestration ----------------
extern "C" void kernel_launch(void* const* d_in, const int* in_sizes, int n_in,
                              void* d_out, int out_size, void* d_ws, size_t ws_size,
                              hipStream_t stream) {
  const int*   X     = (const int*)d_in[0];
  const float* emb   = (const float*)d_in[1];
  const float* U_f   = (const float*)d_in[2];
  const float* V_f   = (const float*)d_in[3];
  const float* b_f   = (const float*)d_in[4];
  const float* U_i   = (const float*)d_in[5];
  const float* V_i   = (const float*)d_in[6];
  const float* b_i   = (const float*)d_in[7];
  const float* U_g   = (const float*)d_in[8];
  const float* V_g   = (const float*)d_in[9];
  const float* b_g   = (const float*)d_in[10];
  const float* U_o   = (const float*)d_in[11];
  const float* U_f1  = (const float*)d_in[12];
  const float* V_f1  = (const float*)d_in[13];
  const float* b_f1  = (const float*)d_in[14];
  const float* U_i1  = (const float*)d_in[15];
  const float* V_i1  = (const float*)d_in[16];
  const float* b_i1  = (const float*)d_in[17];
  const float* U_g1  = (const float*)d_in[18];
  const float* V_g1  = (const float*)d_in[19];
  const float* b_g1  = (const float*)d_in[20];
  const float* U_o1  = (const float*)d_in[21];
  const float* V_o1  = (const float*)d_in[22];
  const float* b_o1  = (const float*)d_in[23];
  const float* W_out = (const float*)d_in[24];
  const float* b_out = (const float*)d_in[25];
  float* out = (float*)d_out;

  char* p = (char*)d_ws;
  auto carve = [&](size_t bytes) -> void* {
    void* r = (void*)p; p += (bytes + 255) & ~(size_t)255; return r;
  };
  unsigned short* Xe   = (unsigned short*)carve((size_t)Bb * Tt * Ee * 2);  // 16 MB
  unsigned short* hbf  = (unsigned short*)carve((size_t)BH * 2);
  unsigned short* h1bf = (unsigned short*)carve((size_t)BH * 2);
  float* c  = (float*)carve((size_t)BH * 4);
  float* c1 = (float*)carve((size_t)BH * 4);
  float* gA = (float*)carve((size_t)4 * BH * 4);
  float* gB = (float*)carve((size_t)4 * BH * 4);
  unsigned short* pU0 = (unsigned short*)carve((size_t)4 * Ee * Hh * 2);
  unsigned short* pV0 = (unsigned short*)carve((size_t)4 * Hh * Hh * 2);
  unsigned short* pU1 = (unsigned short*)carve((size_t)4 * Hh * Hh * 2);
  unsigned short* pV1 = (unsigned short*)carve((size_t)4 * Hh * Hh * 2);
  unsigned short* pW  = (unsigned short*)carve((size_t)Hh * Vv * 2);        // 32.8 MB
  unsigned* bar = (unsigned*)carve(256);

  k_init<<<(BH + TPB - 1) / TPB, TPB, 0, stream>>>(hbf, h1bf, c, c1, bar);
  k_embed<<<2048, TPB, 0, stream>>>(X, emb, Xe);

  struct Job { const float* s; unsigned short* d; int K, N; };
  const Job jobs[] = {
    {U_f,  pU0 + 0 * 131072, Ee, Hh}, {U_i,  pU0 + 1 * 131072, Ee, Hh},
    {U_g,  pU0 + 2 * 131072, Ee, Hh}, {U_o,  pU0 + 3 * 131072, Ee, Hh},
    {V_f,  pV0 + 0 * 262144, Hh, Hh}, {V_i,  pV0 + 1 * 262144, Hh, Hh},
    {V_g,  pV0 + 2 * 262144, Hh, Hh}, {V_o1, pV0 + 3 * 262144, Hh, Hh},  // layer-0 o uses V_o1
    {U_f1, pU1 + 0 * 262144, Hh, Hh}, {U_i1, pU1 + 1 * 262144, Hh, Hh},
    {U_g1, pU1 + 2 * 262144, Hh, Hh}, {U_o1, pU1 + 3 * 262144, Hh, Hh},
    {V_f1, pV1 + 0 * 262144, Hh, Hh}, {V_i1, pV1 + 1 * 262144, Hh, Hh},
    {V_g1, pV1 + 2 * 262144, Hh, Hh}, {V_o1, pV1 + 3 * 262144, Hh, Hh},
    {W_out, pW, Hh, Vv},
  };
  for (const Job& j : jobs) {
    size_t total = (size_t)j.K * j.N;
    int blocks = (int)((total + TPB - 1) / TPB);
    if (blocks > 4096) blocks = 4096;
    k_pack_b<<<blocks, TPB, 0, stream>>>(j.s, j.d, j.K, j.N);
  }

  k_scan<<<NBLK, TPB, 2 * BH * sizeof(unsigned short), stream>>>(   // 128 KB LDS
      Xe, hbf, h1bf, c, c1, gA, gB, pU0, pV0, pU1, pV1,
      b_f, b_i, b_g, b_f1, b_i1, b_g1, b_o1, bar);

  k_head<<<(Vv / 16) / 8, 256, BH * sizeof(unsigned short), stream>>>(  // 64 KB LDS, 250 blocks
      h1bf, pW, b_out, out);

  (void)in_sizes; (void)n_in; (void)out_size; (void)ws_size;
}
